// OSA_Block_6932077215812
// MI455X (gfx1250) — matrix operand order repaired and checked by reading, not compiled, not run on hardware
//
#include <hip/hip_runtime.h>
#include <hip/hip_bf16.h>

typedef _Float16 f16;
typedef __attribute__((ext_vector_type(16))) _Float16 v16h;
typedef __attribute__((ext_vector_type(8)))  float    v8f;

#define WMMA(a,b,c) __builtin_amdgcn_wmma_f32_16x16x32_f16(false,(a),false,(b),(short)0,(c),false,false)

__device__ __forceinline__ float gelu_f(float x){ return 0.5f*x*(1.f + erff(x*0.70710678118654752f)); }
__device__ __forceinline__ float sigmoid_f(float x){ return 1.f/(1.f + __expf(-x)); }

// ---- WMMA fragment loaders (wave32); contiguous k order, same for A and B ----
// A (16x32, MxK) row-major: lane&15 = M row; lanes<16 hold K 0..15, lanes>=16 hold K 16..31
__device__ __forceinline__ v16h ld_a_h(const f16* p, int ld, int lane){
  v16h a; int m = lane & 15; int kb = (lane < 16) ? 0 : 16;
#pragma unroll
  for (int s = 0; s < 16; ++s){ int k = kb + s; a[s] = p[m*ld + k]; }
  return a;
}
__device__ __forceinline__ v16h ld_a_f(const float* p, int ld, int lane){
  v16h a; int m = lane & 15; int kb = (lane < 16) ? 0 : 16;
#pragma unroll
  for (int s = 0; s < 16; ++s){ int k = kb + s; a[s] = (f16)p[m*ld + k]; }
  return a;
}
__device__ __forceinline__ v16h ld_a_h_pad(const f16* p, int ld, int kv, int lane){
  v16h a; int m = lane & 15; int kb = (lane < 16) ? 0 : 16;
#pragma unroll
  for (int s = 0; s < 16; ++s){ int k = kb + s; a[s] = (k < kv) ? p[m*ld + k] : (f16)0.f; }
  return a;
}
// B (32x16, KxN): lane&15 = N col; lanes<16 K rows 0..15, lanes>=16 K rows 16..31
__device__ __forceinline__ v16h ld_b_h(const f16* p, int ld, int lane){
  v16h b; int n = lane & 15; int kb = (lane < 16) ? 0 : 16;
#pragma unroll
  for (int s = 0; s < 16; ++s){ b[s] = p[(kb + s)*ld + n]; }
  return b;
}
// B[k][n] = p[n*ld + k]  (transposed read, e.g. W^T from row-major W)
__device__ __forceinline__ v16h ld_bt_h(const f16* p, int ld, int lane){
  v16h b; int n = lane & 15; int kb = (lane < 16) ? 0 : 16;
#pragma unroll
  for (int s = 0; s < 16; ++s){ b[s] = p[n*ld + kb + s]; }
  return b;
}
__device__ __forceinline__ v16h ld_bt_f(const float* p, int ld, int lane){
  v16h b; int n = lane & 15; int kb = (lane < 16) ? 0 : 16;
#pragma unroll
  for (int s = 0; s < 16; ++s){ b[s] = (f16)p[n*ld + kb + s]; }
  return b;
}
__device__ __forceinline__ v16h ld_bt_h_pad(const f16* p, int ld, int kv, int lane){
  v16h b; int n = lane & 15; int kb = (lane < 16) ? 0 : 16;
#pragma unroll
  for (int s = 0; s < 16; ++s){ int k = kb + s; b[s] = (k < kv) ? p[n*ld + k] : (f16)0.f; }
  return b;
}
// D/C (16x16 f32): lane holds col n=lane&15, rows (lane<16?0:8)+r
__device__ __forceinline__ void st_d_f(float* p, int ld, int lane, v8f d){
  int n = lane & 15, mb = (lane < 16) ? 0 : 8;
#pragma unroll
  for (int r = 0; r < 8; ++r) p[(mb + r)*ld + n] = d[r];
}
__device__ __forceinline__ void st_d_h(f16* p, int ld, int lane, v8f d, float mul){
  int n = lane & 15, mb = (lane < 16) ? 0 : 8;
#pragma unroll
  for (int r = 0; r < 8; ++r) p[(mb + r)*ld + n] = (f16)(d[r] * mul);
}

// =====================================================================
// conv1x1 as GEMM: Y[o,s] = sum_c W[o,c] X[c,s]; Cin fixed 64, 64x64 tile/block
// optional fused LayerNorm2d (channel dim), SE channel scale, bias, GELU, residual
// =====================================================================
template<bool LN2D, bool INSCALE, bool BIAS, bool GELU_ACT, bool RES>
__global__ __launch_bounds__(128)
void conv1x1_k(const float* __restrict__ X, const float* __restrict__ W,
               const float* __restrict__ bias,
               const float* __restrict__ lng, const float* __restrict__ lnb,
               const float* __restrict__ scale,
               const float* __restrict__ R, float* __restrict__ Y, int Cout)
{
  const int HW = 65536;
  const int b  = blockIdx.z;
  const int s0 = blockIdx.x * 64;
  const int o0 = blockIdx.y * 64;
  const int tid = threadIdx.x, lane = tid & 31, wv = tid >> 5;

  __shared__ float Xf[64*65];   // fp32 stage (for LN stats)
  __shared__ f16   Xs[64*72];   // f16 GEMM operand

  for (int i = tid; i < 64*64; i += 128){
    int c = i >> 6, j = i & 63;
    float v = X[(b*64 + c)*HW + s0 + j];
    if (INSCALE) v *= scale[b*64 + c];
    Xf[c*65 + j] = v;
  }
  __syncthreads();
  if (LN2D){
    if (tid < 64){
      int j = tid; float m = 0.f, ss = 0.f;
      for (int c = 0; c < 64; ++c){ float v = Xf[c*65 + j]; m += v; ss += v*v; }
      m *= (1.f/64.f);
      float rs = rsqrtf(ss*(1.f/64.f) - m*m + 1e-6f);
      for (int c = 0; c < 64; ++c)
        Xs[c*72 + j] = (f16)((Xf[c*65 + j] - m)*rs*lng[c] + lnb[c]);
    }
  } else {
    for (int i = tid; i < 64*64; i += 128){ int c = i >> 6, j = i & 63; Xs[c*72 + j] = (f16)Xf[c*65 + j]; }
  }
  __syncthreads();

  const int m0 = wv * 16;
  v16h a0 = ld_a_f(W + (o0 + m0)*64 +  0, 64, lane);   // weights hot in L2 (48KB max)
  v16h a1 = ld_a_f(W + (o0 + m0)*64 + 32, 64, lane);
#pragma unroll
  for (int nt = 0; nt < 4; ++nt){
    v8f acc = {};
    acc = WMMA(a0, ld_b_h(Xs +  0*72 + nt*16, 72, lane), acc);
    acc = WMMA(a1, ld_b_h(Xs + 32*72 + nt*16, 72, lane), acc);
    int n = lane & 15, mb = (lane < 16) ? 0 : 8;
#pragma unroll
    for (int r = 0; r < 8; ++r){
      int o = o0 + m0 + mb + r;
      int j = s0 + nt*16 + n;
      float v = acc[r];
      if (BIAS)     v += bias[o];
      if (GELU_ACT) v  = gelu_f(v);
      if (RES)      v += R[(b*64 + o)*HW + j];
      Y[(b*Cout + o)*HW + j] = v;
    }
  }
}

// =====================================================================
// depthwise 3x3 conv; optional bias/GELU; optional fused SE global-pool
// (block covers one row of one (b,c) slice -> block reduce + 1 FP atomic)
// =====================================================================
template<bool BIAS, bool ACT, bool POOL>
__global__ __launch_bounds__(256)
void dwconv_k(const float* __restrict__ X, const float* __restrict__ W9,
              const float* __restrict__ bias, float* __restrict__ Y, int C,
              float* __restrict__ Pool)
{
  int idx = blockIdx.x * 256 + threadIdx.x;
  int w = idx & 255, h = (idx >> 8) & 255, bc = idx >> 16;
  int c = bc % C;
  const float* xp = X + (size_t)bc * 65536;
  const float* wp = W9 + c*9;
  float acc = 0.f;
#pragma unroll
  for (int di = -1; di <= 1; ++di){
    int hh = h + di; if (hh < 0 || hh > 255) continue;
#pragma unroll
    for (int dj = -1; dj <= 1; ++dj){
      int ww = w + dj; if (ww < 0 || ww > 255) continue;
      acc += wp[(di+1)*3 + (dj+1)] * xp[hh*256 + ww];
    }
  }
  if (BIAS) acc += bias[c];
  if (ACT)  acc = gelu_f(acc);
  Y[idx] = acc;
  if (POOL){
    __shared__ float red[256];
    red[threadIdx.x] = acc; __syncthreads();
    for (int off = 128; off > 0; off >>= 1){
      if (threadIdx.x < off) red[threadIdx.x] += red[threadIdx.x + off];
      __syncthreads();
    }
    if (threadIdx.x == 0) atomicAdd(&Pool[bc], red[0]);
  }
}

// =====================================================================
// fused gff tail: depthwise 3x3 on both 64-ch halves of the 128-ch
// tensor + gelu(y1)*y2 gate, single pass (saves a 134MB write + read)
// =====================================================================
__global__ __launch_bounds__(256)
void dwglu_k(const float* __restrict__ X /*128ch*/, const float* __restrict__ W9 /*128x9*/,
             float* __restrict__ Y /*64ch*/)
{
  int idx = blockIdx.x * 256 + threadIdx.x;   // over 4*64*65536 outputs
  int w = idx & 255, h = (idx >> 8) & 255, bc = idx >> 16;
  int b = bc >> 6, c = bc & 63;
  const float* x1 = X + (size_t)(b*128 +      c) * 65536;
  const float* x2 = X + (size_t)(b*128 + 64 + c) * 65536;
  const float* w1 = W9 + c*9;
  const float* w2 = W9 + (64 + c)*9;
  float a1 = 0.f, a2 = 0.f;
#pragma unroll
  for (int di = -1; di <= 1; ++di){
    int hh = h + di; if (hh < 0 || hh > 255) continue;
#pragma unroll
    for (int dj = -1; dj <= 1; ++dj){
      int ww = w + dj; if (ww < 0 || ww > 255) continue;
      int o = hh*256 + ww, t = (di+1)*3 + (dj+1);
      a1 += w1[t] * x1[o];
      a2 += w2[t] * x2[o];
    }
  }
  Y[idx] = gelu_f(a1) * a2;
}

// ---- SE MLP (pooled sums -> sigmoid(silu(.)W2) channel scales) ----
__global__ __launch_bounds__(64)
void se_k(const float* __restrict__ P, const float* __restrict__ W1,
          const float* __restrict__ W2, float* __restrict__ S)
{
  __shared__ float hid[16];
  int t = threadIdx.x;
  for (int b = 0; b < 4; ++b){
    if (t < 16){
      float a = 0.f;
      for (int c = 0; c < 64; ++c) a += (P[b*64 + c] * (1.f/65536.f)) * W1[t*64 + c];
      hid[t] = a * sigmoid_f(a);          // silu
    }
    __syncthreads();
    float a = 0.f;
    for (int i = 0; i < 16; ++i) a += hid[i] * W2[t*16 + i];
    S[b*64 + t] = sigmoid_f(a);
    __syncthreads();
  }
}

__global__ void zero_k(float* __restrict__ P){ P[threadIdx.x] = 0.f; }

// =====================================================================
// Fully-fused window attention (block or grid). One workgroup = one 8x8
// window of 64 tokens x 64 ch. LN -> QKV GEMM -> per-head softmax attn ->
// out-proj -> +residual, all in LDS. 8 waves, all GEMMs WMMA f16.
// =====================================================================
template<bool GRID>
__global__ __launch_bounds__(256)
void win_attn_k(const float* __restrict__ X, const float* __restrict__ lng,
                const float* __restrict__ lnb, const float* __restrict__ Wqkv,
                const float* __restrict__ Wout, float* __restrict__ Y)
{
  const int HW = 65536;
  const int b  = blockIdx.y;
  const int wid = blockIdx.x;
  const int Xb = wid >> 5, Yb = wid & 31;
  const int tid = threadIdx.x, lane = tid & 31, wv = tid >> 5;

  __shared__ float Sf[64*65];   // raw window, then scores/probs
  __shared__ f16   Th[64*72];   // LN tokens, then attention output O
  __shared__ f16   Qh[64*200];  // q(0..63,*0.25) | k(64..127) | v(128..191)

  auto pix = [&](int t)->int{
    int w1 = t >> 3, w2 = t & 7;
    int h = GRID ? (w1*32 + Xb) : (Xb*8 + w1);
    int w = GRID ? (w2*32 + Yb) : (Yb*8 + w2);
    return h*256 + w;
  };

  for (int i = tid; i < 64*64; i += 256){
    int c = i >> 6, t = i & 63;
    Sf[t*65 + c] = X[(b*64 + c)*HW + pix(t)];
  }
  __syncthreads();
  if (tid < 64){                                  // ln_last over C, eps=1e-5
    int t = tid; float m = 0.f, ss = 0.f;
    for (int c = 0; c < 64; ++c){ float v = Sf[t*65 + c]; m += v; ss += v*v; }
    m *= (1.f/64.f);
    float rs = rsqrtf(ss*(1.f/64.f) - m*m + 1e-5f);
    for (int c = 0; c < 64; ++c)
      Th[t*72 + c] = (f16)((Sf[t*65 + c] - m)*rs*lng[c] + lnb[c]);
  }
  __syncthreads();

  // QKV = T(64x64) @ Wqkv^T(64x192) : 4x12 tiles over 8 waves
  for (int i = 0; i < 6; ++i){
    int tile = wv + 8*i;
    int mt = tile / 12, nt = tile % 12;
    v8f acc = {};
#pragma unroll
    for (int kk = 0; kk < 2; ++kk){
      v16h a  = ld_a_h(Th + mt*16*72 + kk*32, 72, lane);
      v16h bm = ld_bt_f(Wqkv + (nt*16)*64 + kk*32, 64, lane);
      acc = WMMA(a, bm, acc);
    }
    st_d_h(Qh + mt*16*200 + nt*16, 200, lane, acc, (nt < 4) ? 0.25f : 1.0f);  // q *= DH^-0.5
  }
  __syncthreads();

  for (int h = 0; h < 4; ++h){
    // scores = q_h(64x16) @ k_h^T(16x64), K padded 16->32
    for (int i = 0; i < 2; ++i){
      int tile = wv + 8*i;
      int mt = tile >> 2, nt = tile & 3;
      v16h a  = ld_a_h_pad (Qh + mt*16*200 +      h*16, 200, 16, lane);
      v16h bm = ld_bt_h_pad(Qh + nt*16*200 + 64 + h*16, 200, 16, lane);
      v8f acc = {};
      acc = WMMA(a, bm, acc);
      st_d_f(Sf + mt*16*65 + nt*16, 65, lane, acc);
    }
    __syncthreads();
    if (tid < 64){                                // row softmax over 64
      float mx = -1e30f;
      for (int j = 0; j < 64; ++j) mx = fmaxf(mx, Sf[tid*65 + j]);
      float sm = 0.f;
      for (int j = 0; j < 64; ++j){ float e = __expf(Sf[tid*65 + j] - mx); Sf[tid*65 + j] = e; sm += e; }
      float inv = 1.f/sm;
      for (int j = 0; j < 64; ++j) Sf[tid*65 + j] *= inv;
    }
    __syncthreads();
    // O_h = P(64x64) @ v_h(64x16)
    if (wv < 4){
      int mt = wv;
      v8f acc = {};
#pragma unroll
      for (int kk = 0; kk < 2; ++kk){
        v16h a  = ld_a_f(Sf + mt*16*65 + kk*32, 65, lane);
        v16h bm = ld_b_h(Qh + kk*32*200 + 128 + h*16, 200, lane);
        acc = WMMA(a, bm, acc);
      }
      st_d_h(Th + mt*16*72 + h*16, 72, lane, acc, 1.0f);
    }
    __syncthreads();
  }

  // out = O(64x64) @ Wout^T(64x64) + residual(pre-LN x)
  for (int i = 0; i < 2; ++i){
    int tile = wv + 8*i;
    int mt = tile >> 2, nt = tile & 3;
    v8f acc = {};
#pragma unroll
    for (int kk = 0; kk < 2; ++kk){
      v16h a  = ld_a_h(Th + mt*16*72 + kk*32, 72, lane);
      v16h bm = ld_bt_f(Wout + (nt*16)*64 + kk*32, 64, lane);
      acc = WMMA(a, bm, acc);
    }
    int n = lane & 15, mb = (lane < 16) ? 0 : 8;
#pragma unroll
    for (int r = 0; r < 8; ++r){
      int t = mt*16 + mb + r;
      int c = nt*16 + n;
      int gi = (b*64 + c)*HW + pix(t);
      Y[gi] = acc[r] + X[gi];
    }
  }
}

// =====================================================================
// Channel attention core. Per (batch, group, head): q,k,v are 16 x E
// (E=64 non-grid, E=1024 grid). L2-normalize q/k rows, attn = qk^T*temp,
// softmax over 16, O = attn @ v. 4 waves/block, wave = head.
// =====================================================================
template<bool GRID>
__global__ __launch_bounds__(128)
void chan_core_k(const float* __restrict__ QKV, const float* __restrict__ temp,
                 float* __restrict__ O)
{
  const int HW = 65536;
  const int b = blockIdx.y;
  const int g = blockIdx.x;
  const int tid = threadIdx.x, lane = tid & 31, hd = tid >> 5;
  const int E = GRID ? 1024 : 64;

  __shared__ f16 qn[4][16*72];
  __shared__ f16 kn[4][16*72];
  __shared__ f16 vb[4][32*72];      // rows 16..31 zero (K padding)
  __shared__ f16 pr[4][16*32];      // probs, cols 16..31 zero
  __shared__ float invq[4][16];
  __shared__ float invk[4][16];

  auto pix = [&](int e)->int{
    int h, w;
    if (GRID){ int ph = g >> 3, pw = g & 7, eh = e >> 5, ew = e & 31; h = eh*8 + ph; w = ew*8 + pw; }
    else     { int gh = g >> 5, gw = g & 31, ph = e >> 3, pw = e & 7; h = gh*8 + ph; w = gw*8 + pw; }
    return h*256 + w;
  };

  for (int i = lane; i < 16*72; i += 32) vb[hd][16*72 + i] = (f16)0.f;
  for (int i = lane; i < 16*16; i += 32) pr[hd][(i >> 4)*32 + 16 + (i & 15)] = (f16)0.f;

  // phase 1: L2 row norms (lanes 0-15: q rows, lanes 16-31: k rows)
  {
    int row = lane & 15;
    int ch  = (lane < 16) ? (hd*16 + row) : (64 + hd*16 + row);
    const float* base = QKV + (b*192 + ch)*HW;
    float ss = 0.f;
    for (int e = 0; e < E; ++e){ float v = base[pix(e)]; ss += v*v; }
    float inv = 1.f / fmaxf(sqrtf(ss), 1e-12f);
    if (lane < 16) invq[hd][row] = inv; else invk[hd][row] = inv;
  }
  __syncthreads();

  // phase 2: attn(16x16) = qn @ kn^T, accumulated over E in 64-chunks
  v8f att = {};
  for (int c0 = 0; c0 < E; c0 += 64){
    for (int i = lane; i < 16*64; i += 32){
      int row = i >> 6, col = i & 63;
      int pe = pix(c0 + col);
      qn[hd][row*72 + col] = (f16)(QKV[(b*192 +      hd*16 + row)*HW + pe] * invq[hd][row]);
      kn[hd][row*72 + col] = (f16)(QKV[(b*192 + 64 + hd*16 + row)*HW + pe] * invk[hd][row]);
    }
    __syncthreads();
#pragma unroll
    for (int kk = 0; kk < 2; ++kk){
      v16h a  = ld_a_h (qn[hd] + kk*32, 72, lane);
      v16h bm = ld_bt_h(kn[hd] + kk*32, 72, lane);
      att = WMMA(a, bm, att);
    }
    __syncthreads();
  }

  // phase 3: *temp, softmax over 16 cols (cross-lane within 16-lane halves)
  {
    float t = temp[hd];
    v8f p;
#pragma unroll
    for (int r = 0; r < 8; ++r){
      float v = att[r] * t;
      float mx = v;
#pragma unroll
      for (int off = 1; off < 16; off <<= 1) mx = fmaxf(mx, __shfl_xor(mx, off, 32));
      float e = __expf(v - mx);
      float sm = e;
#pragma unroll
      for (int off = 1; off < 16; off <<= 1) sm += __shfl_xor(sm, off, 32);
      p[r] = e / sm;
    }
    st_d_h(pr[hd], 32, lane, p, 1.0f);
  }
  __syncthreads();

  // phase 4: O = probs(16x16 pad 32) @ v(16xE), chunked
  v16h pa = ld_a_h(pr[hd], 32, lane);
  for (int c0 = 0; c0 < E; c0 += 64){
    for (int i = lane; i < 16*64; i += 32){
      int row = i >> 6, col = i & 63;
      vb[hd][row*72 + col] = (f16)QKV[(b*192 + 128 + hd*16 + row)*HW + pix(c0 + col)];
    }
    __syncthreads();
#pragma unroll
    for (int nt = 0; nt < 4; ++nt){
      v16h bm = ld_b_h(vb[hd] + nt*16, 72, lane);
      v8f acc = {};
      acc = WMMA(pa, bm, acc);
      int n = lane & 15, mb = (lane < 16) ? 0 : 8;
#pragma unroll
      for (int r = 0; r < 8; ++r){
        int dc = mb + r;
        int e  = c0 + nt*16 + n;
        O[(b*64 + hd*16 + dc)*HW + pix(e)] = acc[r];
      }
    }
    __syncthreads();
  }
}

// =====================================================================
extern "C" void kernel_launch(void* const* d_in, const int* in_sizes, int n_in,
                              void* d_out, int out_size, void* d_ws, size_t ws_size,
                              hipStream_t stream)
{
  (void)in_sizes; (void)n_in; (void)out_size; (void)ws_size;
  auto F = [&](int i)->const float*{ return (const float*)d_in[i]; };
  const float* x = F(0);

  const size_t NF64  = (size_t)4*64*65536;
  const size_t NF192 = (size_t)4*192*65536;
  float* A    = (float*)d_ws;       // 64ch ping
  float* Bf   = A  + NF64;          // 64ch pong
  float* C1   = Bf + NF64;          // 192ch scratch
  float* D1   = C1 + NF192;         // 192ch scratch
  float* pooled = D1 + NF192;       // [4][64] SE pool sums
  float* sesc   = pooled + 256;     // [4][64] SE scales
  float* OUT  = (float*)d_out;

  // ---- MBConv: gelu(c1) -> gelu(dw)+fused pool -> SE -> c2*scale + x ----
  conv1x1_k<false,false,true,true,false><<<dim3(1024,1,4),128,0,stream>>>(
      x, F(1), F(2), nullptr, nullptr, nullptr, nullptr, C1, 64);
  zero_k<<<1,256,0,stream>>>(pooled);
  dwconv_k<true,true,true><<<(4*64*65536)/256,256,0,stream>>>(C1, F(3), F(4), D1, 64, pooled);
  se_k<<<1,64,0,stream>>>(pooled, F(5), F(6), sesc);
  conv1x1_k<false,true,true,false,true><<<dim3(1024,1,4),128,0,stream>>>(
      D1, F(7), F(8), nullptr, nullptr, sesc, x, A, 64);

  auto gff = [&](const float* in, float* out, int p0){
    // ln2d fused into conv_in tile load; 64 -> 128; fused dw+gate; 64 (+res)
    conv1x1_k<true,false,false,false,false><<<dim3(1024,2,4),128,0,stream>>>(
        in, F(p0+2), nullptr, F(p0), F(p0+1), nullptr, nullptr, C1, 128);
    dwglu_k<<<(4*64*65536)/256,256,0,stream>>>(C1, F(p0+3), D1);
    conv1x1_k<false,false,false,false,true><<<dim3(1024,1,4),128,0,stream>>>(
        D1, F(p0+4), nullptr, nullptr, nullptr, nullptr, in, out, 64);
  };

  auto chan = [&](const float* in, float* out, int p0, bool grid){
    conv1x1_k<true,false,false,false,false><<<dim3(1024,3,4),128,0,stream>>>(
        in, F(p0+2), nullptr, F(p0), F(p0+1), nullptr, nullptr, C1, 192);
    dwconv_k<false,false,false><<<(4*192*65536)/256,256,0,stream>>>(C1, F(p0+3), nullptr, D1, 192, nullptr);
    if (grid) chan_core_k<true ><<<dim3(  64,4),128,0,stream>>>(D1, F(p0+5), C1);
    else      chan_core_k<false><<<dim3(1024,4),128,0,stream>>>(D1, F(p0+5), C1);
    conv1x1_k<false,false,false,false,true><<<dim3(1024,1,4),128,0,stream>>>(
        C1, F(p0+4), nullptr, nullptr, nullptr, nullptr, in, out, 64);
  };

  // ---- attn1 (block window) ----
  win_attn_k<false><<<dim3(1024,4),256,0,stream>>>(A, F(9), F(10), F(11), F(12), Bf);
  gff(Bf, A, 17);                                            // gff1
  chan(A, Bf, 37, false);                                    // channel attn
  gff(Bf, A, 22);                                            // gff2
  win_attn_k<true><<<dim3(1024,4),256,0,stream>>>(A, F(13), F(14), F(15), F(16), Bf); // grid attn
  gff(Bf, A, 27);                                            // gff3
  chan(A, Bf, 43, true);                                     // channel attn (grid)
  gff(Bf, OUT, 32);                                          // gff4 -> d_out
}
